// TinySutra_9655086482198
// MI455X (gfx1250) — matrix operand order repaired and checked by reading, not compiled, not run on hardware
//
#include <hip/hip_runtime.h>

#define D_   256
#define DR_  64
#define M_   2048
#define B_   4
#define T_   8192
#define NC_  10
#define EPS_ 0.3f
#define KTILE 64

typedef __attribute__((ext_vector_type(16))) __bf16 v16bf;
typedef __attribute__((ext_vector_type(8)))  float  v8f;

struct __align__(16) U128 { unsigned int x, y, z, w; };

union Frag16 {
  v16bf v;
  U128  q[2];
};

__device__ __forceinline__ unsigned short f2bf(float x) {
  unsigned int u = __float_as_uint(x);
  u += 0x7FFFu + ((u >> 16) & 1u);   // round-to-nearest-even
  return (unsigned short)(u >> 16);
}

__device__ __forceinline__ float softplusf(float x) {
  return (x > 20.f) ? x : log1pf(__expf(x));
}

// ---------------------------------------------------------------------------
// Generic bf16 WMMA GEMM:  C[M,N] = alpha * A[M,K] @ Bt[N,K]^T + bias[N]
// 256 threads = 8 waves; block tile 32(M) x 128(N); each wave: two 16x16 tiles.
// A staged through double-buffered LDS (2 x 32x64 bf16): one __syncthreads per
// 64-wide K step, next slab's global->LDS copy overlapped with 4 WMMAs.
// ---------------------------------------------------------------------------
__global__ __launch_bounds__(256) void k_gemm(
    const unsigned short* __restrict__ A,  long sAb,
    const unsigned short* __restrict__ Bt, long sBb,
    float* __restrict__ Cf, unsigned short* __restrict__ Cbf, long sCb,
    const float* __restrict__ bias,
    int K, int lda, int ldb, int ldc, float alpha)
{
  __shared__ __align__(16) unsigned short ldsA[2][32 * KTILE];

  const int tid  = threadIdx.x;
  const int wave = tid >> 5;
  const int lane = tid & 31;
  const int tr   = wave & 1;        // 2 row-tiles of 16
  const int tc   = wave >> 1;       // 4 col-groups of 32
  const int hi   = lane >> 4;
  const int ln   = lane & 15;

  const unsigned short* Ab = A  + (long)blockIdx.z * sAb;
  const unsigned short* Bb = Bt + (long)blockIdx.z * sBb;

  const int rowBlk = blockIdx.x * 32;
  const int colBlk = blockIdx.y * 128 + tc * 32;

  v8f acc0 = {0.f,0.f,0.f,0.f,0.f,0.f,0.f,0.f};
  v8f acc1 = {0.f,0.f,0.f,0.f,0.f,0.f,0.f,0.f};

  // B fragment: lane(0-15)=col with K 0..15 ; lane(16-31)=col with K 16..31
  const unsigned short* bp0 = Bb + (long)(colBlk + ln)      * ldb + hi * 16;
  const unsigned short* bp1 = Bb + (long)(colBlk + 16 + ln) * ldb + hi * 16;

  // A tile stage: 8 threads per row, 8 bf16 (16B) each -> 32 x 64 slab
  const int srow = tid >> 3;
  const int scol = (tid & 7) * 8;
  const unsigned short* ap = Ab + (long)(rowBlk + srow) * lda + scol;
  unsigned short* sp0 = &ldsA[0][srow * KTILE + scol];
  unsigned short* sp1 = &ldsA[1][srow * KTILE + scol];

  // preload slab 0
  *(U128*)sp0 = *(const U128*)(ap);
  __syncthreads();

  // A fragment addresses (per-lane K chunk {hi*16..hi*16+15}, +32 for sub1)
  const unsigned short* lf0 = &ldsA[0][(tr * 16 + ln) * KTILE + hi * 16];
  const unsigned short* lf1 = &ldsA[1][(tr * 16 + ln) * KTILE + hi * 16];

  int buf = 0;
  for (int k0 = 0; k0 < K; k0 += KTILE, buf ^= 1) {
    // fetch next A slab into registers (zero slab past the end)
    U128 nxt = {0u, 0u, 0u, 0u};
    if (k0 + KTILE < K) nxt = *(const U128*)(ap + k0 + KTILE);

    __builtin_prefetch((const void*)(bp0 + k0 + KTILE), 0, 1);
    __builtin_prefetch((const void*)(bp1 + k0 + KTILE), 0, 1);

    const unsigned short* lf = buf ? lf1 : lf0;
    Frag16 a0, a1, b00, b01, b10, b11;
    a0.q[0]  = *(const U128*)(lf);
    a0.q[1]  = *(const U128*)(lf + 8);
    a1.q[0]  = *(const U128*)(lf + 32);
    a1.q[1]  = *(const U128*)(lf + 40);
    b00.q[0] = *(const U128*)(bp0 + k0);
    b00.q[1] = *(const U128*)(bp0 + k0 + 8);
    b10.q[0] = *(const U128*)(bp1 + k0);
    b10.q[1] = *(const U128*)(bp1 + k0 + 8);
    b01.q[0] = *(const U128*)(bp0 + k0 + 32);
    b01.q[1] = *(const U128*)(bp0 + k0 + 40);
    b11.q[0] = *(const U128*)(bp1 + k0 + 32);
    b11.q[1] = *(const U128*)(bp1 + k0 + 40);

    acc0 = __builtin_amdgcn_wmma_f32_16x16x32_bf16(
        false, a0.v, false, b00.v, (short)0, acc0, false, false);
    acc1 = __builtin_amdgcn_wmma_f32_16x16x32_bf16(
        false, a0.v, false, b10.v, (short)0, acc1, false, false);
    acc0 = __builtin_amdgcn_wmma_f32_16x16x32_bf16(
        false, a1.v, false, b01.v, (short)0, acc0, false, false);
    acc1 = __builtin_amdgcn_wmma_f32_16x16x32_bf16(
        false, a1.v, false, b11.v, (short)0, acc1, false, false);

    // stage next slab into the other buffer; one barrier per K step
    *(U128*)(buf ? sp0 : sp1) = nxt;
    __syncthreads();
  }

  float*          Cfb = Cf  ? Cf  + (long)blockIdx.z * sCb : (float*)0;
  unsigned short* Cbb = Cbf ? Cbf + (long)blockIdx.z * sCb : (unsigned short*)0;

  const int   row0 = rowBlk + tr * 16 + hi * 8;
  const int   c0   = colBlk + ln;
  const int   c1   = colBlk + 16 + ln;
  const float bv0  = bias ? bias[c0] : 0.0f;
  const float bv1  = bias ? bias[c1] : 0.0f;

  #pragma unroll
  for (int r = 0; r < 8; ++r) {
    const long  o0 = (long)(row0 + r) * ldc + c0;
    const long  o1 = (long)(row0 + r) * ldc + c1;
    const float v0 = acc0[r] * alpha + bv0;
    const float v1 = acc1[r] * alpha + bv1;
    if (Cfb) { Cfb[o0] = v0;        Cfb[o1] = v1; }
    if (Cbb) { Cbb[o0] = f2bf(v0);  Cbb[o1] = f2bf(v1); }
  }
}

// ---------------------------------------------------------------------------
// Embedding gather + chunk-mean (CHUNK=4): h[b,m,:] = mean_c emb[ids[b,4m+c],:]
// ---------------------------------------------------------------------------
__global__ __launch_bounds__(256) void k_embed(
    const int* __restrict__ ids, const float* __restrict__ emb,
    float* __restrict__ h)
{
  const int bm = blockIdx.x;
  const int d  = threadIdx.x;
  const int b  = bm / M_;
  const int m  = bm % M_;
  const int* ir = ids + (long)b * T_ + m * 4;
  float s = 0.f;
  #pragma unroll
  for (int c = 0; c < 4; ++c) s += emb[(long)ir[c] * D_ + d];
  h[(long)bm * D_ + d] = 0.25f * s;
}

// Transpose f32 (R,C) -> bf16 (C,R)
__global__ void k_wT(const float* __restrict__ src,
                     unsigned short* __restrict__ dst, int R, int C) {
  int i = blockIdx.x * blockDim.x + threadIdx.x;
  if (i >= R * C) return;
  int r = i / C, c = i % C;
  dst[(long)c * R + r] = f2bf(src[i]);
}

// conv_w (O,I,K=3) -> Bt (O, 3D) with Bt[o][k*D+i] = w[o,i,k]
__global__ void k_convw(const float* __restrict__ w,
                        unsigned short* __restrict__ dst) {
  int i = blockIdx.x * blockDim.x + threadIdx.x;
  if (i >= D_ * D_ * 3) return;
  int o = i / (D_ * 3);
  int rem = i % (D_ * 3);
  int ii = rem / 3;
  int k  = rem % 3;
  dst[(long)o * (3 * D_) + k * D_ + ii] = f2bf(w[i]);
}

// im2row for conv1d (pad=1): A2[bm][k*D+i] = h[b, m-1+k, i]
__global__ __launch_bounds__(256) void k_im2row(
    const float* __restrict__ h, unsigned short* __restrict__ A2)
{
  long i = (long)blockIdx.x * 256 + threadIdx.x;
  if (i >= (long)B_ * M_ * 3 * D_) return;
  long bm = i / (3 * D_);
  int rem = (int)(i % (3 * D_));
  int k = rem / D_;
  int d = rem % D_;
  int m = (int)(bm % M_);
  int mm = m - 1 + k;
  float v = 0.f;
  if (mm >= 0 && mm < M_) v = h[(bm + (mm - m)) * D_ + d];
  A2[i] = f2bf(v);
}

// st (BM,2D) -> mu, clipped logvar, mu_bf
__global__ __launch_bounds__(256) void k_split(
    const float* __restrict__ st, float* __restrict__ mu,
    float* __restrict__ logvar, unsigned short* __restrict__ mu_bf)
{
  long i = (long)blockIdx.x * 256 + threadIdx.x;
  if (i >= (long)B_ * M_ * D_) return;
  long bm = i / D_; int d = (int)(i % D_);
  float m  = st[bm * (2 * D_) + d];
  float lv = st[bm * (2 * D_) + D_ + d];
  lv = fminf(fmaxf(lv, -4.f), 4.f);
  mu[i] = m; logvar[i] = lv; mu_bf[i] = f2bf(m);
}

// Two simultaneous N=1 projections: o1[bm]=mu.w1+b1, o2[bm]=mu.w2+b2
__global__ __launch_bounds__(256) void k_rowdot2(
    const float* __restrict__ mu,
    const float* __restrict__ w1, const float* __restrict__ b1,
    const float* __restrict__ w2, const float* __restrict__ b2,
    float* __restrict__ o1, float* __restrict__ o2)
{
  __shared__ float r1[256], r2[256];
  const int bm = blockIdx.x, t = threadIdx.x;
  float x = mu[(long)bm * D_ + t];
  r1[t] = x * w1[t];
  r2[t] = x * w2[t];
  __syncthreads();
  for (int s = 128; s > 0; s >>= 1) {
    if (t < s) { r1[t] += r1[t + s]; r2[t] += r2[t + s]; }
    __syncthreads();
  }
  if (t == 0) { o1[bm] = r1[0] + b1[0]; o2[bm] = r2[0] + b2[0]; }
}

// softplus gating + mass-balance normalization -> log_s, log_d
__global__ __launch_bounds__(256) void k_sd(
    const float* __restrict__ rawS, const float* __restrict__ rawD,
    const float* __restrict__ done,
    float* __restrict__ log_s, float* __restrict__ log_d)
{
  __shared__ float r1[256], r2[256];
  __shared__ float cs_sh, cd_sh;
  const int b = blockIdx.x, t = threadIdx.x;
  float ss = 0.f, sd = 0.f;
  for (int m = t; m < M_; m += 256) {
    int i = b * M_ + m;
    float dn = done[i];
    float s = softplusf(rawS[i]) * (0.5f + dn) + 1e-4f;
    float d = softplusf(rawD[i]) * (1.0f - dn) + 1e-4f;
    log_s[i] = s; log_d[i] = d;    // temp storage of raw masses
    ss += s; sd += d;
  }
  r1[t] = ss; r2[t] = sd;
  __syncthreads();
  for (int s = 128; s > 0; s >>= 1) {
    if (t < s) { r1[t] += r1[t + s]; r2[t] += r2[t + s]; }
    __syncthreads();
  }
  if (t == 0) {
    float tot = fminf(r1[0], r2[0]);
    cs_sh = tot / r1[0];
    cd_sh = tot / r2[0];
  }
  __syncthreads();
  const float cs = cs_sh, cd = cd_sh;
  for (int m = t; m < M_; m += 256) {
    int i = b * M_ + m;
    log_s[i] = __logf(log_s[i] * cs + 1e-8f);
    log_d[i] = __logf(log_d[i] * cd + 1e-8f);
  }
}

// u[b,i] = log_d[b,i] - LSE_j( logK[b,i,j] + v[b,j] )   (row sweep, L2-bound)
__global__ __launch_bounds__(256) void k_lse_rows(
    const float* __restrict__ logK, const float* __restrict__ v,
    const float* __restrict__ log_d, float* __restrict__ u)
{
  __shared__ float sm[256], ss[256];
  const int bi = blockIdx.x;               // b*M + i
  const int b  = bi / M_;
  const int t  = threadIdx.x;
  const float* row = logK + (long)bi * M_;
  const float* vb  = v + (long)b * M_;
  float mx = -1e30f, sum = 0.f;
  for (int j = t; j < M_; j += 256) {
    float x = row[j] + vb[j];
    if (x > mx) { sum = sum * __expf(mx - x) + 1.0f; mx = x; }
    else        { sum += __expf(x - mx); }
  }
  sm[t] = mx; ss[t] = sum;
  __syncthreads();
  for (int s = 128; s > 0; s >>= 1) {
    if (t < s) {
      float m1 = sm[t], m2 = sm[t + s];
      float mm = fmaxf(m1, m2);
      ss[t] = ss[t] * __expf(m1 - mm) + ss[t + s] * __expf(m2 - mm);
      sm[t] = mm;
    }
    __syncthreads();
  }
  if (t == 0) u[bi] = log_d[bi] - (sm[0] + __logf(ss[0]));
}

// v[b,j] = log_s[b,j] - LSE_i( logK[b,i,j] + u[b,i] )   (column sweep)
__global__ __launch_bounds__(256) void k_lse_cols(
    const float* __restrict__ logK, const float* __restrict__ u,
    const float* __restrict__ log_s, float* __restrict__ v)
{
  const int b = blockIdx.x;
  const int j = blockIdx.y * 256 + threadIdx.x;
  const float* base = logK + (long)b * M_ * M_ + j;
  const float* ub   = u + (long)b * M_;
  float mx = -1e30f, sum = 0.f;
  for (int i = 0; i < M_; ++i) {
    float x = base[(long)i * M_] + ub[i];
    if (x > mx) { sum = sum * __expf(mx - x) + 1.0f; mx = x; }
    else        { sum += __expf(x - mx); }
  }
  v[b * M_ + j] = log_s[b * M_ + j] - (mx + __logf(sum));
}

// R = exp(logK + u + v), stored bf16 for evidence GEMM
__global__ __launch_bounds__(256) void k_R(
    const float* __restrict__ logK, const float* __restrict__ u,
    const float* __restrict__ v, unsigned short* __restrict__ Rb)
{
  long i = (long)blockIdx.x * 256 + threadIdx.x;
  if (i >= (long)B_ * M_ * M_) return;
  long bi = i / M_;
  int  j  = (int)(i % M_);
  long b  = bi / M_;
  Rb[i] = f2bf(__expf(logK[i] + u[bi] + v[b * M_ + j]));
}

// value (b,M,D) f32 -> valueT (b,D,M) bf16
__global__ __launch_bounds__(256) void k_tval(
    const float* __restrict__ vf, unsigned short* __restrict__ vT)
{
  long i = (long)blockIdx.x * 256 + threadIdx.x;
  if (i >= (long)B_ * D_ * M_) return;
  long b = i / ((long)D_ * M_);
  int rem = (int)(i % ((long)D_ * M_));
  int d = rem / M_;
  int j = rem % M_;
  vT[i] = f2bf(vf[(b * M_ + j) * D_ + d]);
}

// precision-weighted posterior fuse
__global__ __launch_bounds__(256) void k_fuse(
    const float* __restrict__ obs, float* __restrict__ mu,
    float* __restrict__ logvar, unsigned short* __restrict__ mu_bf)
{
  long i = (long)blockIdx.x * 256 + threadIdx.x;
  if (i >= (long)B_ * M_ * D_) return;
  long bm = i / D_; int d = (int)(i % D_);
  float omu = obs[bm * (2 * D_) + d];
  float olv = obs[bm * (2 * D_) + D_ + d];
  olv = fminf(fmaxf(olv, -4.f), 4.f);
  float ps = __expf(-logvar[i]);
  float po = __expf(-olv);
  float den = ps + po + 1e-8f;
  float mn = (ps * mu[i] + po * omu) / den;
  mu[i] = mn; mu_bf[i] = f2bf(mn); logvar[i] = -__logf(den);
}

// advance = sigmoid(mu . w_adv + b); done += (1-done)*advance
__global__ __launch_bounds__(256) void k_advance(
    const float* __restrict__ mu, const float* __restrict__ w,
    const float* __restrict__ b, float* __restrict__ done)
{
  __shared__ float red[256];
  const int bm = blockIdx.x, t = threadIdx.x;
  red[t] = mu[(long)bm * D_ + t] * w[t];
  __syncthreads();
  for (int s = 128; s > 0; s >>= 1) {
    if (t < s) red[t] += red[t + s];
    __syncthreads();
  }
  if (t == 0) {
    float a = 1.f / (1.f + __expf(-(red[0] + b[0])));
    float dn = done[bm];
    done[bm] = dn + (1.f - dn) * a;
  }
}

// logits[b,:] = mean_m(mu[b,m,:]) @ w_head + b_head
__global__ __launch_bounds__(256) void k_head(
    const float* __restrict__ mu, const float* __restrict__ wh,
    const float* __restrict__ bh, float* __restrict__ out)
{
  __shared__ float sh[256];
  const int b = blockIdx.x, t = threadIdx.x;
  float s = 0.f;
  for (int m = 0; m < M_; ++m) s += mu[((long)b * M_ + m) * D_ + t];
  sh[t] = s * (1.0f / M_);
  __syncthreads();
  if (t < NC_) {
    float acc = bh[t];
    for (int d = 0; d < D_; ++d) acc += sh[d] * wh[d * NC_ + t];
    out[b * NC_ + t] = acc;
  }
}

// ---------------------------------------------------------------------------
extern "C" void kernel_launch(void* const* d_in, const int* in_sizes, int n_in,
                              void* d_out, int out_size, void* d_ws, size_t ws_size,
                              hipStream_t stream) {
  (void)in_sizes; (void)n_in; (void)out_size; (void)ws_size;
  const int*   ids      = (const int*)  d_in[0];
  const float* emb      = (const float*)d_in[1];
  const float* conv_w   = (const float*)d_in[2];
  const float* conv_b   = (const float*)d_in[3];
  const float* w_state  = (const float*)d_in[4];
  const float* b_state  = (const float*)d_in[5];
  const float* w_route  = (const float*)d_in[6];
  const float* b_route  = (const float*)d_in[7];
  const float* w_value  = (const float*)d_in[8];
  const float* b_value  = (const float*)d_in[9];
  const float* w_obs    = (const float*)d_in[10];
  const float* b_obs    = (const float*)d_in[11];
  const float* w_supply = (const float*)d_in[12];
  const float* b_supply = (const float*)d_in[13];
  const float* w_demand = (const float*)d_in[14];
  const float* b_demand = (const float*)d_in[15];
  const float* w_adv    = (const float*)d_in[16];
  const float* b_adv    = (const float*)d_in[17];
  const float* w_head   = (const float*)d_in[18];
  const float* b_head   = (const float*)d_in[19];
  float* out = (float*)d_out;

  char* ws = (char*)d_ws;
  size_t off = 0;
  auto alloc = [&](size_t bytes) -> char* {
    char* p = ws + off;
    off = (off + bytes + 255) & ~(size_t)255;
    return p;
  };
  const long BM = (long)B_ * M_;

  float*          h_f    = (float*)alloc(BM * D_ * 4);
  unsigned short* convwT = (unsigned short*)alloc((size_t)D_ * 3 * D_ * 2);
  unsigned short* A2     = (unsigned short*)alloc((size_t)BM * 3 * D_ * 2);
  unsigned short* hconv  = (unsigned short*)alloc((size_t)BM * D_ * 2);
  unsigned short* wstT   = (unsigned short*)alloc((size_t)2 * D_ * D_ * 2);
  unsigned short* wrtT   = (unsigned short*)alloc((size_t)2 * DR_ * D_ * 2);
  unsigned short* wvlT   = (unsigned short*)alloc((size_t)D_ * D_ * 2);
  unsigned short* wobT   = (unsigned short*)alloc((size_t)2 * D_ * D_ * 2);
  float*          proj2  = (float*)alloc(BM * 2 * D_ * 4);     // st / obs
  float*          mu     = (float*)alloc(BM * D_ * 4);
  float*          logvar = (float*)alloc(BM * D_ * 4);
  unsigned short* mu_bf  = (unsigned short*)alloc(BM * D_ * 2);
  unsigned short* rt_bf  = (unsigned short*)alloc(BM * 2 * DR_ * 2);
  float*          rawS   = (float*)alloc(BM * 4);
  float*          rawD   = (float*)alloc(BM * 4);
  float*          log_s  = (float*)alloc(BM * 4);
  float*          log_d  = (float*)alloc(BM * 4);
  float*          u_     = (float*)alloc(BM * 4);
  float*          v_     = (float*)alloc(BM * 4);
  float*          done   = (float*)alloc(BM * 4);
  float*          logK   = (float*)alloc((size_t)B_ * M_ * M_ * 4);
  unsigned short* R_bf   = (unsigned short*)alloc((size_t)B_ * M_ * M_ * 2);
  float*          val_f  = (float*)alloc(BM * D_ * 4);
  unsigned short* valT   = (unsigned short*)alloc((size_t)B_ * D_ * M_ * 2);
  unsigned short* evid   = (unsigned short*)alloc(BM * D_ * 2);

  // ---- stage 0: embedding + weight prep -------------------------------
  k_embed<<<(int)BM, 256, 0, stream>>>(ids, emb, h_f);
  k_convw<<<(D_ * D_ * 3 + 255) / 256, 256, 0, stream>>>(conv_w, convwT);
  k_wT<<<(D_ * 2 * D_ + 255) / 256, 256, 0, stream>>>(w_state, wstT, D_, 2 * D_);
  k_wT<<<(D_ * 2 * DR_ + 255) / 256, 256, 0, stream>>>(w_route, wrtT, D_, 2 * DR_);
  k_wT<<<(D_ * D_ + 255) / 256, 256, 0, stream>>>(w_value, wvlT, D_, D_);
  k_wT<<<(D_ * 2 * D_ + 255) / 256, 256, 0, stream>>>(w_obs, wobT, D_, 2 * D_);

  // ---- conv1d as im2row GEMM (K = 3D = 768), then state projection ----
  k_im2row<<<(int)((BM * 3 * D_ + 255) / 256), 256, 0, stream>>>(h_f, A2);
  k_gemm<<<dim3((int)(BM / 32), D_ / 128, 1), 256, 0, stream>>>(
      A2, 0, convwT, 0, (float*)0, hconv, 0, conv_b,
      3 * D_, 3 * D_, 3 * D_, D_, 1.0f);
  k_gemm<<<dim3((int)(BM / 32), (2 * D_) / 128, 1), 256, 0, stream>>>(
      hconv, 0, wstT, 0, proj2, (unsigned short*)0, 0, b_state,
      D_, D_, D_, 2 * D_, 1.0f);
  k_split<<<(int)((BM * D_ + 255) / 256), 256, 0, stream>>>(proj2, mu, logvar, mu_bf);

  hipMemsetAsync(done, 0, BM * 4, stream);

  const float score_alpha = 1.0f / (8.0f * EPS_);   // 1/sqrt(DR)/EPS

  // ---- 3 belief-update steps -----------------------------------------
  for (int step = 0; step < 3; ++step) {
    // routing projection (s | d), bf16 output for the scores GEMM
    k_gemm<<<dim3((int)(BM / 32), (2 * DR_) / 128, 1), 256, 0, stream>>>(
        mu_bf, 0, wrtT, 0, (float*)0, rt_bf, 0, b_route,
        D_, D_, D_, 2 * DR_, 1.0f);

    // supply / demand scalars + normalization
    k_rowdot2<<<(int)BM, 256, 0, stream>>>(mu, w_supply, b_supply,
                                           w_demand, b_demand, rawS, rawD);
    k_sd<<<B_, 256, 0, stream>>>(rawS, rawD, done, log_s, log_d);

    // logK = (d @ s^T) / (sqrt(DR)*EPS), batched over B, fused scale
    k_gemm<<<dim3(M_ / 32, M_ / 128, B_), 256, 0, stream>>>(
        rt_bf + DR_, (long)M_ * 2 * DR_,       // A = d rows
        rt_bf,       (long)M_ * 2 * DR_,       // Bt = s rows
        logK, (unsigned short*)0, (long)M_ * M_, (const float*)0,
        DR_, 2 * DR_, 2 * DR_, M_, score_alpha);

    // Sinkhorn (8 iterations), L2-resident sweeps over the 64MB logK
    hipMemsetAsync(u_, 0, BM * 4, stream);
    hipMemsetAsync(v_, 0, BM * 4, stream);
    for (int it = 0; it < 8; ++it) {
      k_lse_rows<<<(int)BM, 256, 0, stream>>>(logK, v_, log_d, u_);
      k_lse_cols<<<dim3(B_, M_ / 256), 256, 0, stream>>>(logK, u_, log_s, v_);
    }
    k_R<<<(int)(((long)B_ * M_ * M_ + 255) / 256), 256, 0, stream>>>(logK, u_, v_, R_bf);

    // value projection + transpose to bf16 (D x M) per batch
    k_gemm<<<dim3((int)(BM / 32), D_ / 128, 1), 256, 0, stream>>>(
        mu_bf, 0, wvlT, 0, val_f, (unsigned short*)0, 0, b_value,
        D_, D_, D_, D_, 1.0f);
    k_tval<<<(int)(((long)B_ * D_ * M_ + 255) / 256), 256, 0, stream>>>(val_f, valT);

    // evidence = R @ value  (batched, K = M = 2048), bf16 out
    k_gemm<<<dim3(M_ / 32, D_ / 128, B_), 256, 0, stream>>>(
        R_bf, (long)M_ * M_, valT, (long)D_ * M_,
        (float*)0, evid, (long)M_ * D_, (const float*)0,
        M_, M_, M_, D_, 1.0f);

    // observation projection + Bayesian fuse + gate advance
    k_gemm<<<dim3((int)(BM / 32), (2 * D_) / 128, 1), 256, 0, stream>>>(
        evid, 0, wobT, 0, proj2, (unsigned short*)0, 0, b_obs,
        D_, D_, D_, 2 * D_, 1.0f);
    k_fuse<<<(int)((BM * D_ + 255) / 256), 256, 0, stream>>>(proj2, mu, logvar, mu_bf);
    k_advance<<<(int)BM, 256, 0, stream>>>(mu, w_adv, b_adv, done);
  }

  // ---- head ----------------------------------------------------------
  k_head<<<B_, 256, 0, stream>>>(mu, w_head, b_head, out);
}
